// cnn_layer_71476845740721
// MI455X (gfx1250) — matrix operand order repaired and checked by reading, not compiled, not run on hardware
//
#include <hip/hip_runtime.h>
#include <hip/hip_bf16.h>

typedef __attribute__((ext_vector_type(16))) _Float16 v16h;
typedef __attribute__((ext_vector_type(8)))  float    v8f;

#define EMB   32
#define LSEQ  48
#define NFILT 64
#define OUTC  384          // 6 windows * 64 filters
#define NSLICE 21          // sum of window sizes 1..6

// ---------------- pre-pass 1: embeddings f32 -> f16 table ----------------
__global__ void emb_convert_kernel(const float* __restrict__ emb,
                                   _Float16* __restrict__ emb16) {
  int i = blockIdx.x * blockDim.x + threadIdx.x;
  if (i < 128 * EMB) emb16[i] = (_Float16)emb[i];
}

// ---------------- pre-pass 2: repack conv weights into WMMA B-fragment layout
// For each K-slice s (window w, tap j), n-tile nt (16 filters), the fragment is
// 32 lanes x 16 halfs, lane-contiguous. B layout (16x16x32 f16):
//   lanes 0-15  : column n = lane,   halfs h<8 -> K = h,     h>=8 -> K = 16 + (h-8)
//   lanes 16-31 : column n = lane-16,halfs h<8 -> K = 8 + h, h>=8 -> K = 24 + (h-8)
__global__ void weights_repack_kernel(const float* __restrict__ k1,
                                      const float* __restrict__ k2,
                                      const float* __restrict__ k3,
                                      const float* __restrict__ k4,
                                      const float* __restrict__ k5,
                                      const float* __restrict__ k6,
                                      _Float16* __restrict__ w16) {
  int idx = blockIdx.x * blockDim.x + threadIdx.x;   // ((s*4+nt)*32 + l)*16 + h
  if (idx >= NSLICE * 4 * 32 * 16) return;
  int h  = idx & 15;
  int l  = (idx >> 4) & 31;
  int nt = (idx >> 9) & 3;
  int s  = idx >> 11;
  // decode slice id -> (window w, tap j)
  int rem = s, w = 1;
  while (rem >= w) { rem -= w; ++w; }
  int j = rem;
  const float* kw = (w == 1) ? k1 : (w == 2) ? k2 : (w == 3) ? k3
                  : (w == 4) ? k4 : (w == 5) ? k5 : k6;
  int n = l & 15;
  int khalf = l >> 4;
  int k = (h < 8) ? (khalf * 8 + h) : (16 + khalf * 8 + (h - 8));  // K index within the 32-wide slice
  int f = nt * 16 + n;             // filter (output channel)
  // kernel w layout: (j, e, 1, f) row-major -> ((j*32 + e)*64 + f)
  w16[idx] = (_Float16)kw[(j * EMB + k) * NFILT + f];
}

// ---------------- main kernel: fused gather + conv-as-GEMM + maxpool ------
template <int W>
__device__ inline void conv_task(const _Float16* __restrict__ Alds,
                                 const _Float16* __restrict__ w16,
                                 float* __restrict__ out,
                                 int lane, int nt, int bbase) {
  constexpr int S0 = (W - 1) * W / 2;   // first slice id for window W
  v16h Bf[W];
#pragma unroll
  for (int j = 0; j < W; ++j)
    Bf[j] = *(const v16h*)(w16 + (((S0 + j) * 4 + nt) * 32 + lane) * 16);

  v8f rmax;
#pragma unroll
  for (int q = 0; q < 8; ++q) rmax[q] = -3.0e38f;

  for (int p = 0; p <= LSEQ - W; ++p) {
    v8f acc = {};
#pragma unroll
    for (int j = 0; j < W; ++j) {
      v16h a = *(const v16h*)(Alds + ((p + j) * 32 + lane) * 16);
      acc = __builtin_amdgcn_wmma_f32_16x16x32_f16(
          /*neg_a=*/false, a, /*neg_b=*/false, Bf[j],
          /*c_mod=*/(short)0, acc, /*reuse_a=*/false, /*reuse_b=*/false);
    }
#pragma unroll
    for (int q = 0; q < 8; ++q) rmax[q] = fmaxf(rmax[q], acc[q]);
  }

  // C/D layout: vgpr q, lanes 0-15 -> M=q, N=lane; lanes 16-31 -> M=8+q, N=lane-16
  const int M0  = (lane >> 4) * 8;
  const int N   = lane & 15;
  const int col = (W - 1) * 64 + nt * 16 + N;
#pragma unroll
  for (int q = 0; q < 8; ++q)
    out[(size_t)(bbase + M0 + q) * OUTC + col] = rmax[q];
}

__global__ void __launch_bounds__(256) charcnn_wmma_kernel(
    const int* __restrict__ tokens,
    const _Float16* __restrict__ emb16,
    const _Float16* __restrict__ w16,
    float* __restrict__ out) {
  // A fragments for all 48 positions of this 16-batch tile, WMMA A layout:
  // lane l (r = l&15 row, kh = l>>4): halfs 0-7 = K 8*kh..8*kh+7,
  //                                   halfs 8-15 = K 16+8*kh..16+8*kh+7
  __shared__ __align__(16) _Float16 Alds[LSEQ * 32 * 16];   // 48 KB

  const int tid   = threadIdx.x;
  const int lane  = tid & 31;
  const int wv    = tid >> 5;
  const int bbase = blockIdx.x * 16;

  // fill phase: 48*32 = 1536 lane-slots, 6 per thread
#pragma unroll
  for (int i = 0; i < 6; ++i) {
    int slot = tid + i * 256;
    int p = slot >> 5;
    int l = slot & 31;
    int khalf = l >> 4;
    int tok = tokens[(bbase + (l & 15)) * LSEQ + p];
    const uint4* src = (const uint4*)(emb16 + tok * EMB);  // 4x16B per row
    uint4* dst = (uint4*)(Alds + slot * 16);
    dst[0] = src[khalf];        // halfs 8*kh .. 8*kh+7
    dst[1] = src[2 + khalf];    // halfs 16+8*kh .. 16+8*kh+7
  }
  __syncthreads();

  // compute phase: 24 tasks = (window 1..6) x (4 filter tiles), 3 per wave
#pragma unroll
  for (int it = 0; it < 3; ++it) {
    int t  = wv + it * 8;
    int nt = t & 3;
    switch (t >> 2) {
      case 0: conv_task<1>(Alds, w16, out, lane, nt, bbase); break;
      case 1: conv_task<2>(Alds, w16, out, lane, nt, bbase); break;
      case 2: conv_task<3>(Alds, w16, out, lane, nt, bbase); break;
      case 3: conv_task<4>(Alds, w16, out, lane, nt, bbase); break;
      case 4: conv_task<5>(Alds, w16, out, lane, nt, bbase); break;
      case 5: conv_task<6>(Alds, w16, out, lane, nt, bbase); break;
    }
  }
}

extern "C" void kernel_launch(void* const* d_in, const int* in_sizes, int n_in,
                              void* d_out, int out_size, void* d_ws, size_t ws_size,
                              hipStream_t stream) {
  const int*   tokens = (const int*)d_in[0];
  const float* emb    = (const float*)d_in[1];
  const float* k1     = (const float*)d_in[2];
  const float* k2     = (const float*)d_in[3];
  const float* k3     = (const float*)d_in[4];
  const float* k4     = (const float*)d_in[5];
  const float* k5     = (const float*)d_in[6];
  const float* k6     = (const float*)d_in[7];
  float* out = (float*)d_out;

  // workspace layout: [0, 8192) f16 embedding table, [8192, 8192+86016) B frags
  _Float16* emb16 = (_Float16*)d_ws;
  _Float16* w16   = (_Float16*)((char*)d_ws + 8192);

  emb_convert_kernel<<<16, 256, 0, stream>>>(emb, emb16);
  weights_repack_kernel<<<168, 256, 0, stream>>>(k1, k2, k3, k4, k5, k6, w16);
  charcnn_wmma_kernel<<<8192 / 16, 256, 0, stream>>>(tokens, emb16, w16, out);
}
